// CustomAttention_22668837388613
// MI455X (gfx1250) — matrix operand order repaired and checked by reading, not compiled, not run on hardware
//
#include <hip/hip_runtime.h>

// ---------------------------------------------------------------------------
// Fused LoRA-attention block for MI455X (gfx1250, wave32, WMMA).
// Pipeline:
//   1) x (f32) -> xh (f16)
//   2) wqkv_eff (f16) = qkv_w + lora_A @ lora_B * LORA_SCALE   (rank-8 fold)
//   3) proj_w -> f16
//   4) QKV GEMM (v_wmma_f32_16x16x32_f16, 32x64 tile/wave) with fused
//      per-head LayerNorm epilogue; q scaled by D^-0.5; v stored [B,H,D,N]
//   5) Flash attention: per-wave 16 q-rows, 32-key tiles, online softmax,
//      P staged through LDS to re-shape C-layout -> A-layout, P@V via WMMA
//   6) Output projection GEMM (WMMA, 32x64 tile/wave) + bias -> f32 out
// ---------------------------------------------------------------------------

typedef __attribute__((ext_vector_type(16))) _Float16 v16h;
typedef __attribute__((ext_vector_type(8)))  _Float16 v8h;
typedef __attribute__((ext_vector_type(8)))  float    v8f;

#define DIMC 1024
#define HN   16
#define DH   64
#define NSEQ 2048
#define NB   2
#define NTOK (NB*NSEQ)
#define RK   8

// Build a 16-bit A/B WMMA fragment from two 8-half chunks.
static __device__ __forceinline__ v16h ld16s(const _Float16* p0, const _Float16* p1) {
  v8h lo = *(const v8h*)p0;
  v8h hi = *(const v8h*)p1;
  v16h r;
#pragma unroll
  for (int i = 0; i < 8; ++i) { r[i] = lo[i]; r[i + 8] = hi[i]; }
  return r;
}

static __device__ __forceinline__ v8f wmmaf(v16h a, v16h b, v8f c) {
  return __builtin_amdgcn_wmma_f32_16x16x32_f16(false, a, false, b, (short)0, c,
                                                false, false);
}

// Reductions across a 16-lane half-wave (rows m and m+8 reduced in parallel).
static __device__ __forceinline__ float rsum16(float v) {
  v += __shfl_xor(v, 1, 32);
  v += __shfl_xor(v, 2, 32);
  v += __shfl_xor(v, 4, 32);
  v += __shfl_xor(v, 8, 32);
  return v;
}
static __device__ __forceinline__ float rmax16(float v) {
  v = fmaxf(v, __shfl_xor(v, 1, 32));
  v = fmaxf(v, __shfl_xor(v, 2, 32));
  v = fmaxf(v, __shfl_xor(v, 4, 32));
  v = fmaxf(v, __shfl_xor(v, 8, 32));
  return v;
}

// ---------------------------------------------------------------------------
// Prep kernels
// ---------------------------------------------------------------------------
__global__ void k_cvt_f16(const float* __restrict__ src, _Float16* __restrict__ dst, int n) {
  int i = blockIdx.x * blockDim.x + threadIdx.x;
  if (i < n) dst[i] = (_Float16)src[i];
}

// wqkv_eff[o,c] = qkv_w[o,c] + sum_r A[c,r]*Bm[r,o%DIM]   (LORA_SCALE = 8/8 = 1)
__global__ void k_build_wqkv(const float* __restrict__ qkv_w,
                             const float* __restrict__ Aq, const float* __restrict__ Bq,
                             const float* __restrict__ Ak, const float* __restrict__ Bk,
                             const float* __restrict__ Av, const float* __restrict__ Bv,
                             _Float16* __restrict__ w) {
  int idx = blockIdx.x * blockDim.x + threadIdx.x;
  if (idx >= 3 * DIMC * DIMC) return;
  int o = idx / DIMC, c = idx % DIMC;
  int sel = o >> 10, oc = o & (DIMC - 1);
  const float* A  = (sel == 0) ? Aq : (sel == 1) ? Ak : Av;
  const float* Bm = (sel == 0) ? Bq : (sel == 1) ? Bk : Bv;
  float d = 0.f;
#pragma unroll
  for (int r = 0; r < RK; ++r) d += A[c * RK + r] * Bm[r * DIMC + oc];
  w[idx] = (_Float16)(qkv_w[idx] + d);
}

// ---------------------------------------------------------------------------
// QKV GEMM + LayerNorm epilogue.  grid(NTOK/128, 48), block 128 (4 waves).
// Wave computes a 32(token) x 64(out) tile; col-tile ct selects q/k/v + head.
// ---------------------------------------------------------------------------
__global__ __launch_bounds__(128) void k_qkv_gemm(
    const _Float16* __restrict__ xh, const _Float16* __restrict__ w,
    const float* __restrict__ qn_w, const float* __restrict__ qn_b,
    const float* __restrict__ kn_w, const float* __restrict__ kn_b,
    _Float16* __restrict__ qh, _Float16* __restrict__ kh, _Float16* __restrict__ vt) {
  int wave = threadIdx.x >> 5, lane = threadIdx.x & 31;
  int lh = lane & 15, hi = lane >> 4;
  int rowbase = blockIdx.x * 128 + wave * 32;
  int ct = blockIdx.y, obase = ct * 64;

  const _Float16* aptr0 = xh + (size_t)(rowbase + lh) * DIMC + (hi ? 8 : 0);
  const _Float16* aptr1 = aptr0 + (size_t)16 * DIMC;
  const _Float16* bptr[4];
#pragma unroll
  for (int f = 0; f < 4; ++f)
    bptr[f] = w + (size_t)(obase + f * 16 + lh) * DIMC + (hi ? 8 : 0);

  v8f cc[2][4] = {};
  for (int kb = 0; kb < DIMC; kb += 32) {
    __builtin_prefetch(aptr0 + kb + 128, 0, 1);  // global_prefetch_b8
    __builtin_prefetch(bptr[0] + kb + 128, 0, 1);
    v16h a0 = ld16s(aptr0 + kb, aptr0 + kb + 16);
    v16h a1 = ld16s(aptr1 + kb, aptr1 + kb + 16);
#pragma unroll
    for (int f = 0; f < 4; ++f) {
      v16h b = ld16s(bptr[f] + kb, bptr[f] + kb + 16);
      cc[0][f] = wmmaf(a0, b, cc[0][f]);
      cc[1][f] = wmmaf(a1, b, cc[1][f]);
    }
  }

  int sel = ct >> 4, h = ct & 15;
  int b = rowbase >> 11, n0 = rowbase & (NSEQ - 1);

  if (sel < 2) {  // q or k: per-head LayerNorm over the 64 cols, then store
    const float* lw = sel ? kn_w : qn_w;
    const float* lb = sel ? kn_b : qn_b;
    float wv[4], bv[4];
#pragma unroll
    for (int f = 0; f < 4; ++f) { wv[f] = lw[f * 16 + lh]; bv[f] = lb[f * 16 + lh]; }
    float qscale = sel ? 1.0f : 0.125f;  // D^-0.5 on q
    _Float16* dst = (sel ? kh : qh) + ((size_t)(b * HN + h) * NSEQ + n0) * DH;
#pragma unroll
    for (int mi = 0; mi < 2; ++mi) {
#pragma unroll
      for (int j = 0; j < 8; ++j) {
        float s  = cc[mi][0][j] + cc[mi][1][j] + cc[mi][2][j] + cc[mi][3][j];
        float s2 = cc[mi][0][j]*cc[mi][0][j] + cc[mi][1][j]*cc[mi][1][j] +
                   cc[mi][2][j]*cc[mi][2][j] + cc[mi][3][j]*cc[mi][3][j];
        s = rsum16(s); s2 = rsum16(s2);
        float mu  = s * (1.0f / 64.0f);
        float var = s2 * (1.0f / 64.0f) - mu * mu;
        float rstd = rsqrtf(var + 1e-5f);
        int m = mi * 16 + j + 8 * hi;
#pragma unroll
        for (int f = 0; f < 4; ++f) {
          float y = (cc[mi][f][j] - mu) * rstd * wv[f] + bv[f];
          dst[(size_t)m * DH + f * 16 + lh] = (_Float16)(y * qscale);
        }
      }
    }
  } else {  // v: store transposed [B,H,D,N] so P@V B-frags are contiguous
    _Float16* dst = vt + (size_t)(b * HN + h) * DH * NSEQ;
#pragma unroll
    for (int mi = 0; mi < 2; ++mi) {
#pragma unroll
      for (int f = 0; f < 4; ++f) {
        v8h p;
#pragma unroll
        for (int j = 0; j < 8; ++j) p[j] = (_Float16)cc[mi][f][j];
        *(v8h*)(dst + (size_t)(f * 16 + lh) * NSEQ + n0 + mi * 16 + 8 * hi) = p;
      }
    }
  }
}

// ---------------------------------------------------------------------------
// Flash attention.  grid(NSEQ/64, B*H), block 128 (4 waves, 16 q-rows each).
// ---------------------------------------------------------------------------
__global__ __launch_bounds__(128) void k_attn(
    const _Float16* __restrict__ qh, const _Float16* __restrict__ kh,
    const _Float16* __restrict__ vt, const float* __restrict__ mask,
    _Float16* __restrict__ omid) {
  __shared__ __align__(16) _Float16 pbuf[4][16 * 32];
  int wave = threadIdx.x >> 5, lane = threadIdx.x & 31;
  int lh = lane & 15, hi = lane >> 4;
  int bh = blockIdx.y;
  int qn = blockIdx.x * 64 + wave * 16;

  const _Float16* qb  = qh + (size_t)bh * NSEQ * DH;
  const _Float16* kb_ = kh + (size_t)bh * NSEQ * DH;
  const _Float16* vb  = vt + (size_t)bh * DH * NSEQ;

  // Q A-frags (kept resident): 16 rows x 64 d = two 16x32 frags.
  const _Float16* qp = qb + (size_t)(qn + lh) * DH + (hi ? 8 : 0);
  v16h aq0 = ld16s(qp,      qp + 16);
  v16h aq1 = ld16s(qp + 32, qp + 48);

  v8f oo[4] = {};
  float mrow[8], lrow[8];
#pragma unroll
  for (int j = 0; j < 8; ++j) { mrow[j] = -1e30f; lrow[j] = 0.f; }

  for (int kt = 0; kt < NSEQ; kt += 32) {
    __builtin_prefetch(kb_ + (size_t)(kt + 32 + lh) * DH, 0, 1);
    __builtin_prefetch(vb + (size_t)lh * NSEQ + kt + 32, 0, 1);
    v8f s0 = {}, s1 = {};
#pragma unroll
    for (int kc = 0; kc < 2; ++kc) {
      // K^T B-frags: col = key, K-dim = d (contiguous in kh row)
      const _Float16* kp = kb_ + (size_t)(kt + kc * 16 + lh) * DH + (hi ? 8 : 0);
      v16h bk0 = ld16s(kp,      kp + 16);
      v16h bk1 = ld16s(kp + 32, kp + 48);
      v8f s = {};
      s = wmmaf(aq0, bk0, s);
      s = wmmaf(aq1, bk1, s);
#pragma unroll
      for (int j = 0; j < 8; ++j)
        s[j] += mask[(size_t)(qn + j + 8 * hi) * NSEQ + kt + kc * 16 + lh];
      if (kc == 0) s0 = s; else s1 = s;
    }
    // Online softmax update (per row, via 16-lane reductions).
#pragma unroll
    for (int j = 0; j < 8; ++j) {
      float mx = rmax16(fmaxf(s0[j], s1[j]));
      float nm = fmaxf(mrow[j], mx);
      float alpha = __expf(mrow[j] - nm);
      mrow[j] = nm;
      float p0 = __expf(s0[j] - nm);
      float p1 = __expf(s1[j] - nm);
      lrow[j] = lrow[j] * alpha + rsum16(p0 + p1);
      oo[0][j] *= alpha; oo[1][j] *= alpha; oo[2][j] *= alpha; oo[3][j] *= alpha;
      s0[j] = p0; s1[j] = p1;
    }
    // C-layout P -> LDS (row-major 16x32 f16) -> A-layout frag.
    _Float16* pl = pbuf[wave];
#pragma unroll
    for (int j = 0; j < 8; ++j) {
      int r = j + 8 * hi;
      pl[r * 32 + lh]      = (_Float16)s0[j];
      pl[r * 32 + 16 + lh] = (_Float16)s1[j];
    }
    asm volatile("s_wait_dscnt 0" ::: "memory");
    const _Float16* pr = pl + lh * 32 + (hi ? 8 : 0);
    v16h ap = ld16s(pr, pr + 16);
    // V^T B-frags: col = d, K-dim = key (contiguous in vt row).
#pragma unroll
    for (int f = 0; f < 4; ++f) {
      const _Float16* vp = vb + (size_t)(f * 16 + lh) * NSEQ + kt + (hi ? 8 : 0);
      v16h bv = ld16s(vp, vp + 16);
      oo[f] = wmmaf(ap, bv, oo[f]);
    }
  }

  int h = bh & 15, b = bh >> 4;
#pragma unroll
  for (int j = 0; j < 8; ++j) {
    float inv = 1.0f / lrow[j];
    size_t base = (size_t)(b * NSEQ + qn + j + 8 * hi) * DIMC + h * DH;
#pragma unroll
    for (int f = 0; f < 4; ++f)
      omid[base + f * 16 + lh] = (_Float16)(oo[f][j] * inv);
  }
}

// ---------------------------------------------------------------------------
// Output projection: out = omid @ proj_w.T + b.  grid(NTOK/128, DIMC/64).
// ---------------------------------------------------------------------------
__global__ __launch_bounds__(128) void k_proj(
    const _Float16* __restrict__ omid, const _Float16* __restrict__ w,
    const float* __restrict__ pb, float* __restrict__ out) {
  int wave = threadIdx.x >> 5, lane = threadIdx.x & 31;
  int lh = lane & 15, hi = lane >> 4;
  int rowbase = blockIdx.x * 128 + wave * 32;
  int obase = blockIdx.y * 64;

  const _Float16* aptr0 = omid + (size_t)(rowbase + lh) * DIMC + (hi ? 8 : 0);
  const _Float16* aptr1 = aptr0 + (size_t)16 * DIMC;
  const _Float16* bptr[4];
#pragma unroll
  for (int f = 0; f < 4; ++f)
    bptr[f] = w + (size_t)(obase + f * 16 + lh) * DIMC + (hi ? 8 : 0);

  v8f cc[2][4] = {};
  for (int kb = 0; kb < DIMC; kb += 32) {
    __builtin_prefetch(aptr0 + kb + 128, 0, 1);
    __builtin_prefetch(bptr[0] + kb + 128, 0, 1);
    v16h a0 = ld16s(aptr0 + kb, aptr0 + kb + 16);
    v16h a1 = ld16s(aptr1 + kb, aptr1 + kb + 16);
#pragma unroll
    for (int f = 0; f < 4; ++f) {
      v16h b = ld16s(bptr[f] + kb, bptr[f] + kb + 16);
      cc[0][f] = wmmaf(a0, b, cc[0][f]);
      cc[1][f] = wmmaf(a1, b, cc[1][f]);
    }
  }
#pragma unroll
  for (int mi = 0; mi < 2; ++mi) {
#pragma unroll
    for (int j = 0; j < 8; ++j) {
      size_t t = (size_t)(rowbase + mi * 16 + j + 8 * hi) * DIMC;
#pragma unroll
      for (int f = 0; f < 4; ++f) {
        int o = obase + f * 16 + lh;
        out[t + o] = cc[mi][f][j] + pb[o];
      }
    }
  }
}

// ---------------------------------------------------------------------------
extern "C" void kernel_launch(void* const* d_in, const int* in_sizes, int n_in,
                              void* d_out, int out_size, void* d_ws, size_t ws_size,
                              hipStream_t stream) {
  (void)in_sizes; (void)n_in; (void)out_size; (void)ws_size;
  const float* x      = (const float*)d_in[0];
  const float* mask   = (const float*)d_in[1];
  const float* qkv_w  = (const float*)d_in[2];
  const float* proj_w = (const float*)d_in[3];
  const float* proj_b = (const float*)d_in[4];
  const float* qn_w   = (const float*)d_in[5];
  const float* qn_b   = (const float*)d_in[6];
  const float* kn_w   = (const float*)d_in[7];
  const float* kn_b   = (const float*)d_in[8];
  const float* Aq     = (const float*)d_in[9];
  const float* Bq     = (const float*)d_in[10];
  const float* Ak     = (const float*)d_in[11];
  const float* Bk     = (const float*)d_in[12];
  const float* Av     = (const float*)d_in[13];
  const float* Bv     = (const float*)d_in[14];
  float* out = (float*)d_out;

  char* ws = (char*)d_ws;
  _Float16* xh    = (_Float16*)(ws);                       //  8 MB
  _Float16* wqkv  = (_Float16*)(ws + ((size_t)8  << 20));  //  6 MB
  _Float16* wproj = (_Float16*)(ws + ((size_t)14 << 20));  //  2 MB
  _Float16* qh    = (_Float16*)(ws + ((size_t)16 << 20));  //  8 MB
  _Float16* kh    = (_Float16*)(ws + ((size_t)24 << 20));  //  8 MB
  _Float16* vt    = (_Float16*)(ws + ((size_t)32 << 20));  //  8 MB
  _Float16* omid  = (_Float16*)(ws + ((size_t)40 << 20));  //  8 MB

  k_cvt_f16<<<(NTOK * DIMC + 255) / 256, 256, 0, stream>>>(x, xh, NTOK * DIMC);
  k_build_wqkv<<<(3 * DIMC * DIMC + 255) / 256, 256, 0, stream>>>(
      qkv_w, Aq, Bq, Ak, Bk, Av, Bv, wqkv);
  k_cvt_f16<<<(DIMC * DIMC + 255) / 256, 256, 0, stream>>>(proj_w, wproj, DIMC * DIMC);
  k_qkv_gemm<<<dim3(NTOK / 128, 48), 128, 0, stream>>>(
      xh, wqkv, qn_w, qn_b, kn_w, kn_b, qh, kh, vt);
  k_attn<<<dim3(NSEQ / 64, NB * HN), 128, 0, stream>>>(qh, kh, vt, mask, omid);
  k_proj<<<dim3(NTOK / 128, DIMC / 64), 128, 0, stream>>>(omid, wproj, proj_b, out);
}
